// MOE_42331197670160
// MI455X (gfx1250) — matrix operand order repaired and checked by reading, not compiled, not run on hardware
//
#include <hip/hip_runtime.h>

#define D_IN   1024
#define D_H    2048
#define D_OUT  1024
#define NEXP   8
#define TOPK   2
#define NBATCH 2048
#define NPAIR  (NBATCH * TOPK)   // 4096 total (token, expert) pairs, always

typedef __attribute__((ext_vector_type(16))) __bf16 v16bf;
typedef __attribute__((ext_vector_type(8)))  float  v8f;
typedef __attribute__((ext_vector_type(4)))  unsigned int v4u;

union FragBF { v16bf v; v4u q[2]; };

__device__ __forceinline__ unsigned short f32_bf16(float f) {
  unsigned u = __float_as_uint(f);
  u += 0x7FFFu + ((u >> 16) & 1u);   // round-to-nearest-even
  return (unsigned short)(u >> 16);
}

// -------------------- 0: init --------------------
__global__ void init_kernel(int* __restrict__ counts) {
  if (threadIdx.x < NEXP) counts[threadIdx.x] = 0;
}

// -------------------- 1: gating + top-2 --------------------
__global__ __launch_bounds__(256) void gate_topk_kernel(
    const float* __restrict__ x, const float* __restrict__ Wg,
    const float* __restrict__ bg,
    float* __restrict__ topw, int* __restrict__ topidx, int* __restrict__ counts) {
  int t = blockIdx.x * 256 + threadIdx.x;
  if (t >= NBATCH) return;
  float acc[NEXP];
#pragma unroll
  for (int e = 0; e < NEXP; ++e) acc[e] = bg[e];
  const float* xr = x + (size_t)t * D_IN;
  for (int d = 0; d < D_IN; ++d) {
    float xv = xr[d];
    const float4* w = (const float4*)(Wg + d * NEXP);
    float4 w0 = w[0], w1 = w[1];
    acc[0] = fmaf(xv, w0.x, acc[0]); acc[1] = fmaf(xv, w0.y, acc[1]);
    acc[2] = fmaf(xv, w0.z, acc[2]); acc[3] = fmaf(xv, w0.w, acc[3]);
    acc[4] = fmaf(xv, w1.x, acc[4]); acc[5] = fmaf(xv, w1.y, acc[5]);
    acc[6] = fmaf(xv, w1.z, acc[6]); acc[7] = fmaf(xv, w1.w, acc[7]);
  }
  float m = acc[0];
#pragma unroll
  for (int e = 1; e < NEXP; ++e) m = fmaxf(m, acc[e]);
  float p[NEXP]; float s = 0.f;
#pragma unroll
  for (int e = 0; e < NEXP; ++e) { p[e] = expf(acc[e] - m); s += p[e]; }
  float inv = 1.f / s;
#pragma unroll
  for (int e = 0; e < NEXP; ++e) p[e] *= inv;
  // top-2 (strict > keeps first index on ties, matching jax.lax.top_k)
  int i1 = 0; float v1 = p[0];
#pragma unroll
  for (int e = 1; e < NEXP; ++e) if (p[e] > v1) { v1 = p[e]; i1 = e; }
  int i2 = (i1 == 0) ? 1 : 0; float v2 = p[i2];
#pragma unroll
  for (int e = 0; e < NEXP; ++e)
    if (e != i1 && e != i2 && p[e] > v2) { v2 = p[e]; i2 = e; }
  topw[t * TOPK + 0] = v1;  topidx[t * TOPK + 0] = i1;
  topw[t * TOPK + 1] = v2;  topidx[t * TOPK + 1] = i2;
  atomicAdd(&counts[i1], 1);
  atomicAdd(&counts[i2], 1);
}

// -------------------- 2: prefix over 8 counts --------------------
__global__ void prefix_kernel(const int* __restrict__ counts,
                              int* __restrict__ offs, int* __restrict__ cursor) {
  if (threadIdx.x == 0) {
    int s = 0;
#pragma unroll
    for (int e = 0; e < NEXP; ++e) { offs[e] = s; s += counts[e]; cursor[e] = 0; }
  }
}

// -------------------- 3: scatter tokens into expert-grouped slots ----------
__global__ __launch_bounds__(256) void scatter_kernel(
    const int* __restrict__ topidx, int* __restrict__ cursor,
    const int* __restrict__ offs, int* __restrict__ pair_token,
    int* __restrict__ slot_of) {
  int t = blockIdx.x * 256 + threadIdx.x;
  if (t >= NBATCH) return;
#pragma unroll
  for (int k = 0; k < TOPK; ++k) {
    int e = topidx[t * TOPK + k];
    int slot = offs[e] + atomicAdd(&cursor[e], 1);
    pair_token[slot] = t;
    slot_of[t * TOPK + k] = slot;
  }
}

// -------------------- 4: GEMM1  h = relu(x_gathered @ W1 + b1) ------------
// 128 (slots) x 128 (h cols) tile per block; 8 waves -> 4x2 wave grid,
// each wave owns 2x4 WMMA 16x16 f32 accumulators. K streamed 32-wide.
#define SSTRIDE 40  // padded LDS row stride (bf16 elems): bank-conflict free
__global__ __launch_bounds__(256) void expert_gemm1_kernel(
    const float* __restrict__ x, const float* __restrict__ W1,
    const float* __restrict__ b1,
    const int* __restrict__ counts, const int* __restrict__ offs,
    const int* __restrict__ pair_token, unsigned short* __restrict__ h_ws) {
  __shared__ __align__(16) unsigned short sA[128 * SSTRIDE];
  __shared__ __align__(16) unsigned short sB[128 * SSTRIDE];
  __shared__ int sTok[128];

  const int e  = blockIdx.y >> 4;
  const int mt = blockIdx.y & 15;
  const int cnt = counts[e];
  if (mt * 128 >= cnt) return;
  const int mstart = offs[e] + mt * 128;
  const int mend   = offs[e] + cnt;
  const int n0 = blockIdx.x * 128;

  const int tid  = threadIdx.x;
  const int lane = tid & 31;
  const int wave = tid >> 5;
  const int lrow = lane & 15;
  const int lhi  = lane >> 4;
  const int wm = wave & 3;    // rows 32*wm
  const int wn = wave >> 2;   // cols 64*wn

  if (tid < 128) {
    int slot = mstart + tid;
    sTok[tid] = (slot < mend) ? pair_token[slot] : 0;
  }
  __syncthreads();
  const int myrow  = tid >> 1;
  const int mypart = tid & 1;
  const float* xsrc = x + (size_t)sTok[myrow] * D_IN + mypart * 16;

  v8f acc[2][4];
#pragma unroll
  for (int i = 0; i < 2; ++i)
#pragma unroll
    for (int t = 0; t < 4; ++t) acc[i][t] = {};

  for (int k0 = 0; k0 < D_IN; k0 += 32) {
    // stage A: gathered x rows, cols [k0, k0+32), row-major bf16
    {
      unsigned short* dst = sA + myrow * SSTRIDE + mypart * 16;
      const float4* s4 = (const float4*)(xsrc + k0);
#pragma unroll
      for (int j = 0; j < 4; ++j) {
        float4 v = s4[j];
        dst[4 * j + 0] = f32_bf16(v.x); dst[4 * j + 1] = f32_bf16(v.y);
        dst[4 * j + 2] = f32_bf16(v.z); dst[4 * j + 3] = f32_bf16(v.w);
      }
    }
    // stage B: W1[e][k0+d][n0+cc] -> col-major sB[cc][d], float4 wide loads
#pragma unroll
    for (int j = 0; j < 4; ++j) {
      int pos = (tid + 256 * j) * 4;  // 0..4092, step 4
      int d  = pos >> 7;              // 0..31
      int cc = pos & 127;             // multiple of 4
      float4 v = *(const float4*)(W1 + ((size_t)e * D_IN + (k0 + d)) * D_H + n0 + cc);
      sB[(cc + 0) * SSTRIDE + d] = f32_bf16(v.x);
      sB[(cc + 1) * SSTRIDE + d] = f32_bf16(v.y);
      sB[(cc + 2) * SSTRIDE + d] = f32_bf16(v.z);
      sB[(cc + 3) * SSTRIDE + d] = f32_bf16(v.w);
    }
    __syncthreads();
    FragBF af[2];
#pragma unroll
    for (int i = 0; i < 2; ++i) {
      const unsigned short* ap = sA + (32 * wm + 16 * i + lrow) * SSTRIDE + 8 * lhi;
      af[i].q[0] = *(const v4u*)ap;
      af[i].q[1] = *(const v4u*)(ap + 16);
    }
#pragma unroll
    for (int t = 0; t < 4; ++t) {
      FragBF bf;
      const unsigned short* bp = sB + (64 * wn + 16 * t + lrow) * SSTRIDE + 8 * lhi;
      bf.q[0] = *(const v4u*)bp;
      bf.q[1] = *(const v4u*)(bp + 16);
#pragma unroll
      for (int i = 0; i < 2; ++i)
        acc[i][t] = __builtin_amdgcn_wmma_f32_16x16x32_bf16(
            false, af[i].v, false, bf.v, (short)0, acc[i][t], false, false);
    }
    __syncthreads();
  }

  // epilogue: +b1, relu, store bf16 h (fast path for full tiles)
  const bool full = (mstart + 128) <= mend;
#pragma unroll
  for (int t = 0; t < 4; ++t) {
    int hcol = n0 + 64 * wn + 16 * t + lrow;
    float b1v = b1[(size_t)e * D_H + hcol];
    if (full) {
#pragma unroll
      for (int i = 0; i < 2; ++i)
#pragma unroll
        for (int r = 0; r < 8; ++r) {
          int slot = mstart + 32 * wm + 16 * i + r + 8 * lhi;
          float v = fmaxf(acc[i][t][r] + b1v, 0.f);
          h_ws[(size_t)slot * D_H + hcol] = f32_bf16(v);
        }
    } else {
#pragma unroll
      for (int i = 0; i < 2; ++i)
#pragma unroll
        for (int r = 0; r < 8; ++r) {
          int slot = mstart + 32 * wm + 16 * i + r + 8 * lhi;
          if (slot < mend) {
            float v = fmaxf(acc[i][t][r] + b1v, 0.f);
            h_ws[(size_t)slot * D_H + hcol] = f32_bf16(v);
          }
        }
    }
  }
}

// -------------------- 5: GEMM2  y = h @ W2 + b2 ---------------------------
// A staging (h is already bf16) uses CDNA5 async memory->LDS copies tracked
// by ASYNCcnt, overlapped with the f32->bf16 conversion staging of W2.
__global__ __launch_bounds__(256) void expert_gemm2_kernel(
    const unsigned short* __restrict__ h_ws, const float* __restrict__ W2,
    const float* __restrict__ b2,
    const int* __restrict__ counts, const int* __restrict__ offs,
    float* __restrict__ y_ws) {
  __shared__ __align__(16) unsigned short sA[128 * SSTRIDE];
  __shared__ __align__(16) unsigned short sB[128 * SSTRIDE];

  const int e  = blockIdx.y >> 4;
  const int mt = blockIdx.y & 15;
  const int cnt = counts[e];
  if (mt * 128 >= cnt) return;
  const int mstart = offs[e] + mt * 128;
  const int mend   = offs[e] + cnt;
  const int n0 = blockIdx.x * 128;

  const int tid  = threadIdx.x;
  const int lane = tid & 31;
  const int wave = tid >> 5;
  const int lrow = lane & 15;
  const int lhi  = lane >> 4;
  const int wm = wave & 3;
  const int wn = wave >> 2;

  const int myrow  = tid >> 1;
  const int mypart = tid & 1;
  int myslot = mstart + myrow;
  myslot = (myslot < NPAIR - 1) ? myslot : (NPAIR - 1);  // clamp pad rows
  const unsigned short* hsrc = h_ws + (size_t)myslot * D_H + mypart * 16;
  // LDS byte address of this thread's A-tile destination (low 32 bits of the
  // generic pointer are the LDS offset per the ISA aperture mapping).
  unsigned lds_a = (unsigned)(unsigned long long)(sA + myrow * SSTRIDE + mypart * 16);

  v8f acc[2][4];
#pragma unroll
  for (int i = 0; i < 2; ++i)
#pragma unroll
    for (int t = 0; t < 4; ++t) acc[i][t] = {};

  for (int k0 = 0; k0 < D_H; k0 += 32) {
    // A stage: two async 16B copies h_ws -> LDS (no VGPR round trip, ASYNCcnt)
    asm volatile("global_load_async_to_lds_b128 %0, %1, off"
                 :: "v"(lds_a), "v"(hsrc + k0) : "memory");
    asm volatile("global_load_async_to_lds_b128 %0, %1, off"
                 :: "v"(lds_a + 16), "v"(hsrc + k0 + 8) : "memory");
    // B stage: W2[e][k0+d][n0+cc] -> col-major sB[cc][d], float4 wide loads
#pragma unroll
    for (int j = 0; j < 4; ++j) {
      int pos = (tid + 256 * j) * 4;
      int d  = pos >> 7;
      int cc = pos & 127;
      float4 v = *(const float4*)(W2 + ((size_t)e * D_H + (k0 + d)) * D_OUT + n0 + cc);
      sB[(cc + 0) * SSTRIDE + d] = f32_bf16(v.x);
      sB[(cc + 1) * SSTRIDE + d] = f32_bf16(v.y);
      sB[(cc + 2) * SSTRIDE + d] = f32_bf16(v.z);
      sB[(cc + 3) * SSTRIDE + d] = f32_bf16(v.w);
    }
    asm volatile("s_wait_asynccnt 0x0" ::: "memory");
    __syncthreads();
    FragBF af[2];
#pragma unroll
    for (int i = 0; i < 2; ++i) {
      const unsigned short* ap = sA + (32 * wm + 16 * i + lrow) * SSTRIDE + 8 * lhi;
      af[i].q[0] = *(const v4u*)ap;
      af[i].q[1] = *(const v4u*)(ap + 16);
    }
#pragma unroll
    for (int t = 0; t < 4; ++t) {
      FragBF bf;
      const unsigned short* bp = sB + (64 * wn + 16 * t + lrow) * SSTRIDE + 8 * lhi;
      bf.q[0] = *(const v4u*)bp;
      bf.q[1] = *(const v4u*)(bp + 16);
#pragma unroll
      for (int i = 0; i < 2; ++i)
        acc[i][t] = __builtin_amdgcn_wmma_f32_16x16x32_bf16(
            false, af[i].v, false, bf.v, (short)0, acc[i][t], false, false);
    }
    __syncthreads();
  }

  const bool full = (mstart + 128) <= mend;
#pragma unroll
  for (int t = 0; t < 4; ++t) {
    int ocol = n0 + 64 * wn + 16 * t + lrow;
    float b2v = b2[(size_t)e * D_OUT + ocol];
    if (full) {
#pragma unroll
      for (int i = 0; i < 2; ++i)
#pragma unroll
        for (int r = 0; r < 8; ++r) {
          int slot = mstart + 32 * wm + 16 * i + r + 8 * lhi;
          y_ws[(size_t)slot * D_OUT + ocol] = acc[i][t][r] + b2v;
        }
    } else {
#pragma unroll
      for (int i = 0; i < 2; ++i)
#pragma unroll
        for (int r = 0; r < 8; ++r) {
          int slot = mstart + 32 * wm + 16 * i + r + 8 * lhi;
          if (slot < mend)
            y_ws[(size_t)slot * D_OUT + ocol] = acc[i][t][r] + b2v;
        }
    }
  }
}

// -------------------- 6: combine  out = w0*y[s0] + w1*y[s1] ---------------
__global__ __launch_bounds__(256) void combine_kernel(
    const float* __restrict__ y_ws, const float* __restrict__ topw,
    const int* __restrict__ slot_of, float* __restrict__ out) {
  int idx = blockIdx.x * 256 + threadIdx.x;   // float4 index
  int token = idx >> 8;                       // 256 float4 per row
  int col = (idx & 255) * 4;
  float w0 = topw[token * TOPK + 0], w1 = topw[token * TOPK + 1];
  int   s0 = slot_of[token * TOPK + 0], s1 = slot_of[token * TOPK + 1];
  float4 a = *(const float4*)(y_ws + (size_t)s0 * D_OUT + col);
  float4 b = *(const float4*)(y_ws + (size_t)s1 * D_OUT + col);
  float4 o;
  o.x = fmaf(w0, a.x, w1 * b.x);
  o.y = fmaf(w0, a.y, w1 * b.y);
  o.z = fmaf(w0, a.z, w1 * b.z);
  o.w = fmaf(w0, a.w, w1 * b.w);
  *(float4*)(out + (size_t)token * D_OUT + col) = o;
}

extern "C" void kernel_launch(void* const* d_in, const int* in_sizes, int n_in,
                              void* d_out, int out_size, void* d_ws, size_t ws_size,
                              hipStream_t stream) {
  (void)in_sizes; (void)n_in; (void)out_size; (void)ws_size;
  const float* x  = (const float*)d_in[0];
  const float* Wg = (const float*)d_in[1];
  const float* bg = (const float*)d_in[2];
  const float* W1 = (const float*)d_in[3];
  const float* b1 = (const float*)d_in[4];
  const float* W2 = (const float*)d_in[5];
  const float* b2 = (const float*)d_in[6];
  float* out = (float*)d_out;

  char* ws = (char*)d_ws;
  float* topw     = (float*)(ws + 0);            // 16 KB
  int*   topidx   = (int*)  (ws + 16384);        // 16 KB
  int*   counts   = (int*)  (ws + 32768);        // 32 B
  int*   cursor   = (int*)  (ws + 32800);        // 32 B
  int*   offs     = (int*)  (ws + 32832);        // 32 B
  int*   pair_tok = (int*)  (ws + 32864);        // 16 KB
  int*   slot_of  = (int*)  (ws + 49248);        // 16 KB
  unsigned short* h_ws = (unsigned short*)(ws + 65792);               // 16 MB bf16
  float* y_ws = (float*)(ws + 65792 + (size_t)NPAIR * D_H * 2);       // 16 MB f32

  init_kernel<<<1, 32, 0, stream>>>(counts);
  gate_topk_kernel<<<NBATCH / 256, 256, 0, stream>>>(x, Wg, bg, topw, topidx, counts);
  prefix_kernel<<<1, 1, 0, stream>>>(counts, offs, cursor);
  scatter_kernel<<<NBATCH / 256, 256, 0, stream>>>(topidx, cursor, offs, pair_tok, slot_of);
  expert_gemm1_kernel<<<dim3(D_H / 128, NEXP * 16), 256, 0, stream>>>(
      x, W1, b1, counts, offs, pair_tok, h_ws);
  expert_gemm2_kernel<<<dim3(D_OUT / 128, NEXP * 16), 256, 0, stream>>>(
      h_ws, W2, b2, counts, offs, y_ws);
  combine_kernel<<<(NBATCH * D_OUT / 4) / 256, 256, 0, stream>>>(y_ws, topw, slot_of, out);
}
